// RPEMultiHeadAttention_61993557951037
// MI455X (gfx1250) — compile-verified, matrix operand order checked by reading, not run on hardware
//
#include <hip/hip_runtime.h>
#include <hip/hip_bf16.h>
#include <math.h>

#define C_DIM 256
#define N_SEQ 512
#define H_HEADS 8
#define D_HEAD 32
#define SCALE_ATT 0.17677669529663687f  // 1/sqrt(32)

typedef __attribute__((ext_vector_type(16))) __bf16 bf16x16;
typedef __attribute__((ext_vector_type(8)))  __bf16 bf16x8;
typedef __attribute__((ext_vector_type(8)))  float  f32x8;

// ---------------------------------------------------------------------------
// WMMA helper: D = A(16x32 bf16) * B(32x16 bf16) + C(16x16 f32)
// ---------------------------------------------------------------------------
__device__ __forceinline__ f32x8 wmma_bf16(bf16x16 a, bf16x16 b, f32x8 c) {
  return __builtin_amdgcn_wmma_f32_16x16x32_bf16(false, a, false, b, (short)0, c,
                                                 false, false);
}

// A/B fragment (lane,element)->K map (wave32, 16-bit 16x32 layout, ISA 7.12.2):
// lane L: idx16 = L&15 (M for A, N for B), g = L>>4
// elem e (0..7)  -> k = k0 + g*8 + e
// elem e (8..15) -> k = k0 + 16 + g*8 + (e-8)

__device__ __forceinline__ bf16x16 frag_from_f32(const float* __restrict__ row,
                                                 int k0, int g) {
  const float* p0 = row + k0 + g * 8;
  const float* p1 = p0 + 16;
  bf16x16 r;
#pragma unroll
  for (int i = 0; i < 8; ++i) r[i] = (__bf16)p0[i];
#pragma unroll
  for (int i = 0; i < 8; ++i) r[8 + i] = (__bf16)p1[i];
  return r;
}

// Non-temporal variant: used for the one-shot 537MB embed stream (keep L2 for K/V/W)
__device__ __forceinline__ bf16x16 frag_from_f32_nt(const float* __restrict__ row,
                                                    int k0, int g) {
  const float* p0 = row + k0 + g * 8;
  const float* p1 = p0 + 16;
  bf16x16 r;
#pragma unroll
  for (int i = 0; i < 8; ++i) r[i] = (__bf16)__builtin_nontemporal_load(p0 + i);
#pragma unroll
  for (int i = 0; i < 8; ++i) r[8 + i] = (__bf16)__builtin_nontemporal_load(p1 + i);
  return r;
}

__device__ __forceinline__ bf16x16 frag_from_bf16(const __bf16* __restrict__ row,
                                                  int k0, int g) {
  bf16x8 lo = *(const bf16x8*)(row + k0 + g * 8);        // 16B load
  bf16x8 hi = *(const bf16x8*)(row + k0 + 16 + g * 8);   // 16B load
  bf16x16 r;
#pragma unroll
  for (int i = 0; i < 8; ++i) { r[i] = lo[i]; r[8 + i] = hi[i]; }
  return r;
}

// ---------------------------------------------------------------------------
// K0: transpose+convert weights.  WT[w][n][k] = bf16(W[k][n]) for Wq,Wk,Wv,Wout;
//     WpT[k][c] = Wp[c][k] (f32, for coalesced u computation)
// ---------------------------------------------------------------------------
__global__ void __launch_bounds__(256)
prep_weights(const float* __restrict__ wq, const float* __restrict__ wk,
             const float* __restrict__ wv, const float* __restrict__ wout,
             const float* __restrict__ wp, __bf16* __restrict__ wt,
             float* __restrict__ wpt) {
  const int w = blockIdx.y;   // 0..4
  const int n = blockIdx.x;   // 0..255
  const int t = threadIdx.x;  // 0..255
  if (w < 4) {
    const float* src = (w == 0) ? wq : (w == 1) ? wk : (w == 2) ? wv : wout;
    wt[(size_t)w * 65536 + n * 256 + t] = (__bf16)src[t * 256 + n];
  } else {
    wpt[n * 256 + t] = wp[t * 256 + n];
  }
}

// ---------------------------------------------------------------------------
// K1: q/k/v projections.  Y(1024x256) = X(1024x256) @ W.  blockIdx.y selects.
// 8 waves/WG, one 16x16 tile per wave, K-loop 8x WMMA.
// ---------------------------------------------------------------------------
__global__ void __launch_bounds__(256)
proj_kernel(const float* __restrict__ xq, const float* __restrict__ xk,
            const float* __restrict__ xv, const __bf16* __restrict__ wt,
            float* __restrict__ qproj, __bf16* __restrict__ kproj,
            float* __restrict__ vproj) {
  const int which = blockIdx.y;
  const float* X = (which == 0) ? xq : (which == 1) ? xk : xv;
  const __bf16* W = wt + (size_t)which * 65536;  // WT[n][k]
  const int tid = threadIdx.x, lane = tid & 31, wave = tid >> 5;
  const int l15 = lane & 15, g = lane >> 4;
  const int tile = blockIdx.x * 8 + wave;             // 0..1023
  const int m0 = (tile >> 4) * 16, n0 = (tile & 15) * 16;
  const float* xrow = X + (size_t)(m0 + l15) * C_DIM;
  const __bf16* wrow = W + (size_t)(n0 + l15) * C_DIM;
  f32x8 acc = {0.f, 0.f, 0.f, 0.f, 0.f, 0.f, 0.f, 0.f};
#pragma unroll
  for (int kt = 0; kt < 8; ++kt)
    acc = wmma_bf16(frag_from_f32(xrow, kt * 32, g),
                    frag_from_bf16(wrow, kt * 32, g), acc);
  const int nn = n0 + l15;
#pragma unroll
  for (int r = 0; r < 8; ++r) {
    const int mm = m0 + g * 8 + r;
    const float v = acc[r];
    if (which == 0)      qproj[(size_t)mm * C_DIM + nn] = v;
    else if (which == 1) kproj[(size_t)mm * C_DIM + nn] = (__bf16)v;
    else                 vproj[(size_t)mm * C_DIM + nn] = v;
  }
}

// ---------------------------------------------------------------------------
// K2: u_ext[b,n][h][0:256]   = Σ_d Wp[c][h*32+d] q[b,n][h*32+d]   (scores_p coeffs)
//     u_ext[b,n][h][256:512] = q[b,n][c] masked to head h          (scores_e coeffs)
// ---------------------------------------------------------------------------
__global__ void __launch_bounds__(256)
uext_kernel(const float* __restrict__ qproj, const float* __restrict__ wpt,
            __bf16* __restrict__ uext) {
  const int row = blockIdx.x;    // b*512+n
  const int c = threadIdx.x;     // 0..255
  __shared__ float qrow[256];
  qrow[c] = qproj[(size_t)row * 256 + c];
  __syncthreads();
  __bf16* u = uext + (size_t)row * 4096;
  for (int h = 0; h < H_HEADS; ++h) {
    float acc = 0.f;
#pragma unroll
    for (int d = 0; d < D_HEAD; ++d)
      acc = fmaf(wpt[(h * 32 + d) * 256 + c], qrow[h * 32 + d], acc);
    u[h * 512 + c] = (__bf16)acc;
    u[h * 512 + 256 + c] = ((c >> 5) == h) ? (__bf16)qrow[c] : (__bf16)0.f;
  }
}

// ---------------------------------------------------------------------------
// K3: fused attention core.  One WG per (b,n).
//   S[m,h] = [embed_row | kproj_row] . u_ext[h]   (WMMA, K=512, bf16)
//   softmax over m per head, then hidden[c] = Σ_m attn[h(c),m] * vproj[m,c]
// ---------------------------------------------------------------------------
__global__ void __launch_bounds__(256)
attn_kernel(const float* __restrict__ embed, const __bf16* __restrict__ kproj,
            const __bf16* __restrict__ uext, const float* __restrict__ vproj,
            float* __restrict__ hidden) {
  const int n = blockIdx.x, b = blockIdx.y;
  const int tid = threadIdx.x, lane = tid & 31, wave = tid >> 5;
  const int l15 = lane & 15, g = lane >> 4;

  __shared__ __align__(16) __bf16 u_lds[16 * 512];  // rows 8..15 zero-padded
  __shared__ float s_lds[8 * 512];                  // scores -> attn

  const __bf16* usrc = uext + (size_t)(b * N_SEQ + n) * (H_HEADS * 512);
  for (int i = tid; i < 8 * 512; i += 256) u_lds[i] = usrc[i];
  const __bf16 z = (__bf16)0.f;
  for (int i = tid; i < 8 * 512; i += 256) u_lds[8 * 512 + i] = z;
  __syncthreads();

  const float* Eb = embed + (size_t)(b * N_SEQ + n) * N_SEQ * C_DIM;
  const __bf16* Kb = kproj + (size_t)b * N_SEQ * C_DIM;
  const __bf16* urow = u_lds + l15 * 512;  // B fragment row (h = l15)

  for (int mt = wave; mt < 32; mt += 8) {   // uniform across block: EXEC all-1s
    const int m0 = mt * 16;
    const float* erow = Eb + (size_t)(m0 + l15) * C_DIM;
    const __bf16* krow = Kb + (size_t)(m0 + l15) * C_DIM;
    f32x8 acc = {0.f, 0.f, 0.f, 0.f, 0.f, 0.f, 0.f, 0.f};
#pragma unroll
    for (int kt = 0; kt < 8; ++kt)   // embed part (K 0..255), NT stream
      acc = wmma_bf16(frag_from_f32_nt(erow, kt * 32, g),
                      frag_from_bf16(urow, kt * 32, g), acc);
#pragma unroll
    for (int kt = 0; kt < 8; ++kt)   // k-proj part (K 256..511), L2-hot
      acc = wmma_bf16(frag_from_bf16(krow, kt * 32, g),
                      frag_from_bf16(urow, 256 + kt * 32, g), acc);
    if (l15 < H_HEADS) {
#pragma unroll
      for (int r = 0; r < 8; ++r)
        s_lds[l15 * 512 + m0 + g * 8 + r] = acc[r] * SCALE_ATT;
    }
  }
  __syncthreads();

  {  // softmax: wave w owns head w
    const int h = wave;
    float mx = -3.0e38f;
    for (int m = lane; m < 512; m += 32) mx = fmaxf(mx, s_lds[h * 512 + m]);
#pragma unroll
    for (int off = 16; off > 0; off >>= 1) mx = fmaxf(mx, __shfl_xor(mx, off, 32));
    float sum = 0.f;
    for (int m = lane; m < 512; m += 32) {
      float e = __expf(s_lds[h * 512 + m] - mx);
      s_lds[h * 512 + m] = e;
      sum += e;
    }
#pragma unroll
    for (int off = 16; off > 0; off >>= 1) sum += __shfl_xor(sum, off, 32);
    const float inv = 1.f / sum;
    for (int m = lane; m < 512; m += 32) s_lds[h * 512 + m] *= inv;
  }
  __syncthreads();

  {  // attn @ V: thread c; LDS broadcast of attn, coalesced V (L2-resident)
    const int c = tid, h = c >> 5;
    const float* V = vproj + (size_t)b * N_SEQ * C_DIM + c;
    const float* arow = s_lds + h * 512;
    float acc = 0.f;
#pragma unroll 4
    for (int m = 0; m < 512; ++m) acc = fmaf(arow[m], V[(size_t)m * C_DIM], acc);
    hidden[(size_t)(b * N_SEQ + n) * C_DIM + c] = acc;
  }
}

// ---------------------------------------------------------------------------
// K4: out = hidden @ Wout + bout
// ---------------------------------------------------------------------------
__global__ void __launch_bounds__(256)
out_kernel(const float* __restrict__ hidden, const __bf16* __restrict__ wt_out,
           const float* __restrict__ bout, float* __restrict__ out) {
  const int tid = threadIdx.x, lane = tid & 31, wave = tid >> 5;
  const int l15 = lane & 15, g = lane >> 4;
  const int tile = blockIdx.x * 8 + wave;
  const int m0 = (tile >> 4) * 16, n0 = (tile & 15) * 16;
  const float* xrow = hidden + (size_t)(m0 + l15) * C_DIM;
  const __bf16* wrow = wt_out + (size_t)(n0 + l15) * C_DIM;
  f32x8 acc = {0.f, 0.f, 0.f, 0.f, 0.f, 0.f, 0.f, 0.f};
#pragma unroll
  for (int kt = 0; kt < 8; ++kt)
    acc = wmma_bf16(frag_from_f32(xrow, kt * 32, g),
                    frag_from_bf16(wrow, kt * 32, g), acc);
  const int nn = n0 + l15;
  const float bias = bout[nn];
#pragma unroll
  for (int r = 0; r < 8; ++r)
    out[(size_t)(m0 + g * 8 + r) * C_DIM + nn] = acc[r] + bias;
}

// ---------------------------------------------------------------------------
extern "C" void kernel_launch(void* const* d_in, const int* in_sizes, int n_in,
                              void* d_out, int out_size, void* d_ws, size_t ws_size,
                              hipStream_t stream) {
  const float* input_q = (const float*)d_in[0];
  const float* input_k = (const float*)d_in[1];
  const float* input_v = (const float*)d_in[2];
  const float* embed   = (const float*)d_in[3];
  const float* Wq      = (const float*)d_in[4];
  const float* Wk      = (const float*)d_in[5];
  const float* Wv      = (const float*)d_in[6];
  const float* Wp      = (const float*)d_in[7];
  const float* Wout    = (const float*)d_in[8];
  const float* bout    = (const float*)d_in[9];
  float* out = (float*)d_out;

  char* ws = (char*)d_ws;
  __bf16* WT     = (__bf16*)(ws);                     // 4x256x256 bf16 = 512KB
  float*  WpT    = (float*)(ws + (512 << 10));        // 256KB
  float*  qproj  = (float*)(ws + (768 << 10));        // 1MB
  float*  vproj  = (float*)(ws + (1792 << 10));       // 1MB
  __bf16* kproj  = (__bf16*)(ws + (2816 << 10));      // 512KB
  __bf16* uext   = (__bf16*)(ws + (3328 << 10));      // 8MB
  float*  hidden = (float*)(ws + (11520 << 10));      // 1MB  (total ~12.3MB)

  prep_weights<<<dim3(256, 5), 256, 0, stream>>>(Wq, Wk, Wv, Wout, Wp, WT, WpT);
  proj_kernel<<<dim3(128, 3), 256, 0, stream>>>(input_q, input_k, input_v, WT,
                                                qproj, kproj, vproj);
  uext_kernel<<<dim3(1024), 256, 0, stream>>>(qproj, WpT, uext);
  attn_kernel<<<dim3(512, 2), 256, 0, stream>>>(embed, kproj, uext, vproj, hidden);
  out_kernel<<<dim3(128), 256, 0, stream>>>(hidden, WT + 3 * 65536, bout, out);
}
